// AdvSeqDiscriminatorS4_54511724921193
// MI455X (gfx1250) — matrix operand order repaired and checked by reading, not compiled, not run on hardware
//
#include <hip/hip_runtime.h>

typedef __attribute__((ext_vector_type(16))) __bf16 v16bf;
typedef __attribute__((ext_vector_type(8)))  float  v8f;

#define Hdim 1024
#define Ldim 1024
#define Bdim 8
#define Mrows (Bdim * Ldim)      // 8192
#define N2s  32
#define NSSM 6
#define Rdim 64                  // H/16

// ---------- helpers ----------
static __device__ __forceinline__ unsigned short f2bf(float f) {
  unsigned int u = __float_as_uint(f);
  u += 0x7FFFu + ((u >> 16) & 1u);        // round-to-nearest-even
  return (unsigned short)(u >> 16);
}
static __device__ __forceinline__ float sigmoidf_(float x) {
  return 1.0f / (1.0f + expf(-x));
}
static __device__ __forceinline__ float gelu_tanh(float y) {
  float t = 0.7978845608028654f * (y + 0.044715f * y * y * y);
  return 0.5f * y * (1.0f + tanhf(t));
}

// ---------- cast fp32 -> bf16 ----------
__global__ void k_cast_bf16(const float* __restrict__ src,
                            unsigned short* __restrict__ dst, int n) {
  int i = blockIdx.x * blockDim.x + threadIdx.x;
  if (i < n) dst[i] = f2bf(src[i]);
}

// ---------- conv weight (O,I,K) -> bf16 fragment-major B  ----------
// dst[ktile][ntile][lane][e], ktile over (ksz*H)/32, ntile over H/16.
// element (kk, o): lane = ((kk>>4)&1)*16 | (o&15), e = kk&15.
__global__ void k_prep_convw(const float* __restrict__ w,
                             unsigned short* __restrict__ dst, int ksz, int n) {
  int idx = blockIdx.x * blockDim.x + threadIdx.x;
  if (idx >= n) return;
  const int o = idx & 1023;
  const int kk = idx >> 10;          // 0 .. ksz*1024-1
  const int i = kk & 1023;
  const int kpos = kk >> 10;
  const float v = w[((long)o * Hdim + i) * ksz + kpos];
  const int ktile = kk >> 5;
  const int lane = (((kk >> 4) & 1) << 4) | (o & 15);
  const int e = kk & 15;
  dst[(((long)ktile * (Hdim >> 4) + (o >> 4)) << 9) + (lane << 4) + e] = f2bf(v);
}

// ---------- ow (2H,H) -> bf16 fragment-major B (K=H, N=2H) ----------
__global__ void k_prep_ow(const float* __restrict__ ow,
                          unsigned short* __restrict__ dst, int n) {
  int idx = blockIdx.x * blockDim.x + threadIdx.x;
  if (idx >= n) return;
  const int o = idx & 2047;
  const int hh = idx >> 11;
  const float v = ow[(long)o * Hdim + hh];
  const int ktile = hh >> 5;
  const int lane = (((hh >> 4) & 1) << 4) | (o & 15);
  const int e = hh & 15;
  dst[(((long)ktile * (2 * Hdim >> 4) + (o >> 4)) << 9) + (lane << 4) + e] = f2bf(v);
}

// ---------- WMMA GEMM (implicit im2col over l-shift) ----------
// A: (Mrows, 1024) bf16 (b,l,h).  Bp: fragment-major (Kd/32, N/16, 32, 16) bf16.
// Block tile 32x256: 8 waves as 2(M) x 4(N); each wave -> 4 accumulators (16x64).
__global__ __launch_bounds__(256) void k_gemm_bf16(
    const unsigned short* __restrict__ A,
    const unsigned short* __restrict__ Bp,
    const float* __restrict__ bias,
    float* __restrict__ C,
    int N, int ksz) {
  __shared__ __align__(16) unsigned short sA[2 * 32 * 16];  // 32x32 bf16 swizzled

  const int tid  = threadIdx.x;
  const int wave = tid >> 5;
  const int lane = tid & 31;
  const int hi   = lane >> 4;
  const int ln   = lane & 15;
  const int mw   = wave & 1;          // M half of block tile
  const int nw   = wave >> 1;         // N quarter of block tile
  const int m0   = blockIdx.y << 5;
  const int ntile0  = (blockIdx.x << 4) + (nw << 2);
  const int ntilesN = N >> 4;
  const int ctr  = ksz >> 1;

  // --- A staging role: thread stages 4 bf16 (uint2) of a 32x32 tile ---
  const int r  = tid >> 3;            // tile row 0..31
  const int kb = (tid & 7) << 2;      // k base 0,4,...,28
  const int arow = m0 + r;
  const int ab = arow >> 10;
  const int al = arow & 1023;
  const int lane_a = (r & 15) | (((kb >> 4) & 1) << 4);
  const int e_a = kb & 15;                                // multiple of 4
  const int sw = ((((r >> 4) << 5 | lane_a) << 4) + e_a) >> 2;  // uint2 index
  // --- A fragment base for this wave (uint4 index) ---
  const int fa_base = ((mw << 5) | lane) << 1;

  v8f z = {0.f, 0.f, 0.f, 0.f, 0.f, 0.f, 0.f, 0.f};
  v8f acc0 = z, acc1 = z, acc2 = z, acc3 = z;

  const int ktmax = ksz << 5;         // Kd/32

  for (int kpos = 0; kpos < ksz; ++kpos) {
    const int lsrc = al + kpos - ctr;
    const bool av = (lsrc >= 0) && (lsrc < Ldim);
    const long abase = (long)((ab << 10) | (lsrc & 1023)) * Hdim;
    for (int i0 = 0; i0 < Hdim; i0 += 32) {
      // stage A tile (swizzled): 1 x b64 global load + 1 x b64 LDS store
      uint2 aval = make_uint2(0u, 0u);
      if (av) aval = *(const uint2*)(A + abase + i0 + kb);
      ((uint2*)sA)[sw] = aval;
      __syncthreads();

      // A fragment: 2 x ds_load_b128, contiguous per lane
      union { uint4 q[2]; v16bf v; } fa;
      fa.q[0] = ((const uint4*)sA)[fa_base];
      fa.q[1] = ((const uint4*)sA)[fa_base + 1];

      // B fragments: fragment-major global, 2 x global_load_b128 per tile
      const long kt = ((long)kpos << 5) + (i0 >> 5);
      const uint4* bp =
          (const uint4*)(Bp + ((kt * ntilesN + ntile0) << 9) + (lane << 4));
      union { uint4 q[2]; v16bf v; } fb;
      fb.q[0] = bp[0];   fb.q[1] = bp[1];
      acc0 = __builtin_amdgcn_wmma_f32_16x16x32_bf16(
          false, fa.v, false, fb.v, (short)0, acc0, false, false);
      fb.q[0] = bp[64];  fb.q[1] = bp[65];
      acc1 = __builtin_amdgcn_wmma_f32_16x16x32_bf16(
          false, fa.v, false, fb.v, (short)0, acc1, false, false);
      fb.q[0] = bp[128]; fb.q[1] = bp[129];
      acc2 = __builtin_amdgcn_wmma_f32_16x16x32_bf16(
          false, fa.v, false, fb.v, (short)0, acc2, false, false);
      fb.q[0] = bp[192]; fb.q[1] = bp[193];
      acc3 = __builtin_amdgcn_wmma_f32_16x16x32_bf16(
          false, fa.v, false, fb.v, (short)0, acc3, false, false);

      if (kt + 1 < ktmax)
        __builtin_prefetch((const void*)(bp + ((long)ntilesN << 6)), 0, 0);
      __syncthreads();
    }
  }

  // epilogue: bias + store (per VGPR row, 16 consecutive cols per half-wave)
  const int mrow0 = m0 + (mw << 4) + (hi << 3);
  {
    union { float f[8]; v8f v; } r0, r1, r2, r3;
    r0.v = acc0; r1.v = acc1; r2.v = acc2; r3.v = acc3;
#pragma unroll
    for (int j = 0; j < 4; ++j) {
      const int col = ((ntile0 + j) << 4) + ln;
      const float bb = bias[col];
      const float* fp = (j == 0) ? r0.f : (j == 1) ? r1.f : (j == 2) ? r2.f : r3.f;
#pragma unroll
      for (int q = 0; q < 8; ++q) {
        C[(long)(mrow0 + q) * N + col] = fp[q] + bb;
      }
    }
  }
}

// ---------- channel LayerNorm over H (per (b,l) row) + ReLU ----------
__global__ __launch_bounds__(256) void k_cln_relu(
    const float* __restrict__ in, const float* __restrict__ g,
    const float* __restrict__ be, float* __restrict__ out) {
  const int row = blockIdx.x;
  const int t = threadIdx.x;
  const float* p = in + (long)row * Hdim;
  __shared__ float r1[256], r2[256];
  float v[4], s = 0.f, sq = 0.f;
#pragma unroll
  for (int j = 0; j < 4; ++j) {
    v[j] = p[t + j * 256];
    s += v[j];
    sq += v[j] * v[j];
  }
  r1[t] = s; r2[t] = sq;
  __syncthreads();
  for (int st = 128; st > 0; st >>= 1) {
    if (t < st) { r1[t] += r1[t + st]; r2[t] += r2[t + st]; }
    __syncthreads();
  }
  const float mean = r1[0] * (1.0f / Hdim);
  const float var = r2[0] * (1.0f / Hdim) - mean * mean;
  const float rstd = rsqrtf(var + 1e-5f);
#pragma unroll
  for (int j = 0; j < 4; ++j) {
    const int c = t + j * 256;
    float o = (v[j] - mean) * rstd * g[c] + be[c];
    out[(long)row * Hdim + c] = fmaxf(o, 0.0f);
  }
}

// ---------- CBAM: per (b,h) avg/max over L ----------
__global__ void k_cbam_pool(const float* __restrict__ h,
                            float* __restrict__ avg, float* __restrict__ mx) {
  const int idx = blockIdx.x * blockDim.x + threadIdx.x;  // B*H
  if (idx >= Bdim * Hdim) return;
  const int b = idx >> 10, c = idx & 1023;
  float s = 0.f, m = -1e30f;
  for (int l = 0; l < Ldim; ++l) {
    float x = h[((long)b * Ldim + l) * Hdim + c];
    s += x;
    m = fmaxf(m, x);
  }
  avg[idx] = s * (1.0f / Ldim);
  mx[idx] = m;
}

// ---------- CBAM: shared MLP, channel gate ----------
__global__ __launch_bounds__(64) void k_cbam_mlp(
    const float* __restrict__ avg, const float* __restrict__ mx,
    const float* __restrict__ w1, const float* __restrict__ b1,
    const float* __restrict__ w2, const float* __restrict__ b2,
    float* __restrict__ gate) {
  const int b = blockIdx.x;
  const int t = threadIdx.x;  // 0..63 = r
  __shared__ float hidA[Rdim], hidM[Rdim];
  float dA = b1[t], dM = b1[t];
  for (int c = 0; c < Hdim; ++c) {
    const float w = w1[(long)t * Hdim + c];
    dA += w * avg[b * Hdim + c];
    dM += w * mx[b * Hdim + c];
  }
  hidA[t] = fmaxf(dA, 0.f);
  hidM[t] = fmaxf(dM, 0.f);
  __syncthreads();
  for (int c = t; c < Hdim; c += 64) {
    float sA = 0.f, sM = 0.f;
    for (int r = 0; r < Rdim; ++r) {
      const float w = w2[(long)c * Rdim + r];
      sA += hidA[r] * w;
      sM += hidM[r] * w;
    }
    gate[b * Hdim + c] = sigmoidf_(sA + sM + 2.0f * b2[c]);
  }
}

// ---------- CBAM: apply channel gate, per-(b,l) mean/max over H ----------
__global__ __launch_bounds__(256) void k_cbam_scale_sp(
    float* __restrict__ h, const float* __restrict__ gate,
    float* __restrict__ spmean, float* __restrict__ spmax) {
  const int row = blockIdx.x;  // (b,l)
  const int b = row >> 10;
  const int t = threadIdx.x;
  __shared__ float r1[256], r2[256];
  float s = 0.f, m = -1e30f;
#pragma unroll
  for (int j = 0; j < 4; ++j) {
    const int c = t + j * 256;
    float x = h[(long)row * Hdim + c] * gate[b * Hdim + c];
    h[(long)row * Hdim + c] = x;
    s += x;
    m = fmaxf(m, x);
  }
  r1[t] = s; r2[t] = m;
  __syncthreads();
  for (int st = 128; st > 0; st >>= 1) {
    if (t < st) { r1[t] += r1[t + st]; r2[t] = fmaxf(r2[t], r2[t + st]); }
    __syncthreads();
  }
  if (t == 0) {
    spmean[row] = r1[0] * (1.0f / Hdim);
    spmax[row] = r2[0];
  }
}

// ---------- CBAM: spatial 7-tap conv + sigmoid + scale ----------
__global__ __launch_bounds__(256) void k_cbam_spatial(
    float* __restrict__ h, const float* __restrict__ spmean,
    const float* __restrict__ spmax, const float* __restrict__ spw,
    const float* __restrict__ spb) {
  const int row = blockIdx.x;
  const int b = row >> 10, l = row & 1023;
  float acc = spb[0];
#pragma unroll
  for (int k = 0; k < 7; ++k) {
    const int ls = l + k - 3;
    if (ls >= 0 && ls < Ldim) {
      acc += spw[k] * spmean[b * Ldim + ls] + spw[7 + k] * spmax[b * Ldim + ls];
    }
  }
  const float att = sigmoidf_(acc);
  const int t = threadIdx.x;
#pragma unroll
  for (int j = 0; j < 4; ++j) h[(long)row * Hdim + t + j * 256] *= att;
}

// ---------- S4D coefficient prep: w = exp(dt*A), C' = C*(w-1)/A ----------
__global__ void k_s4_coef(const float* __restrict__ log_dt,
                          const float* __restrict__ logA,
                          const float* __restrict__ Aim,
                          const float* __restrict__ Cre,
                          const float* __restrict__ Cim,
                          float* __restrict__ Wr, float* __restrict__ Wi,
                          float* __restrict__ Cr, float* __restrict__ Ci) {
  const int idx = blockIdx.x * blockDim.x + threadIdx.x;  // H*N2
  if (idx >= Hdim * N2s) return;
  const int hch = idx >> 5;
  const float dt = expf(log_dt[hch]);
  const float Are = -expf(logA[idx]);
  const float Aimv = Aim[idx];
  const float ar = Are * dt, ai = Aimv * dt;
  const float er = expf(ar);
  const float wr = er * cosf(ai);
  const float wi = er * sinf(ai);
  const float e1r = wr - 1.0f, e1i = wi;               // exp(dtA)-1
  const float den = Are * Are + Aimv * Aimv;
  const float tr = (e1r * Are + e1i * Aimv) / den;     // (exp(dtA)-1)/A
  const float ti = (e1i * Are - e1r * Aimv) / den;
  const float cre = Cre[idx], cim = Cim[idx];
  Wr[idx] = wr;
  Wi[idx] = wi;
  Cr[idx] = cre * tr - cim * ti;
  Ci[idx] = cre * ti + cim * tr;
}

// ---------- S4D diagonal recurrence + skip + GELU, emit bf16 ----------
__global__ __launch_bounds__(256) void k_s4_scan(
    const float* __restrict__ u, const float* __restrict__ Wr,
    const float* __restrict__ Wi, const float* __restrict__ Cr,
    const float* __restrict__ Ci, const float* __restrict__ D,
    unsigned short* __restrict__ out_bf) {
  const int idx = blockIdx.x * blockDim.x + threadIdx.x;  // B*H
  if (idx >= Bdim * Hdim) return;
  const int b = idx >> 10, hch = idx & 1023;
  float wr[N2s], wi[N2s], cr[N2s], ci[N2s], sr[N2s], si[N2s];
#pragma unroll
  for (int n = 0; n < N2s; ++n) {
    wr[n] = Wr[hch * N2s + n];
    wi[n] = Wi[hch * N2s + n];
    cr[n] = Cr[hch * N2s + n];
    ci[n] = Ci[hch * N2s + n];
    sr[n] = 0.f;
    si[n] = 0.f;
  }
  const float Dh = D[hch];
  const float* up = u + (long)b * Ldim * Hdim + hch;
  unsigned short* op = out_bf + (long)b * Ldim * Hdim + hch;
  for (int l = 0; l < Ldim; ++l) {
    const float uv = up[(long)l * Hdim];
    float acc = 0.f;
#pragma unroll
    for (int n = 0; n < N2s; ++n) {
      const float nsr = wr[n] * sr[n] - wi[n] * si[n] + uv;
      const float nsi = wr[n] * si[n] + wi[n] * sr[n];
      sr[n] = nsr;
      si[n] = nsi;
      acc += cr[n] * nsr - ci[n] * nsi;
    }
    const float y = 2.0f * acc + Dh * uv;
    op[(long)l * Hdim] = f2bf(gelu_tanh(y));
  }
}

// ---------- GLU: a * sigmoid(gate) ----------
__global__ void k_glu(const float* __restrict__ y2, float* __restrict__ out, int n) {
  const int idx = blockIdx.x * blockDim.x + threadIdx.x;  // Mrows*H
  if (idx >= n) return;
  const int row = idx >> 10, c = idx & 1023;
  const float a = y2[(long)row * 2048 + c];
  const float gt = y2[(long)row * 2048 + 1024 + c];
  out[idx] = a * sigmoidf_(gt);
}

// ---------- attention pooling: per-row score & fc dot ----------
__global__ __launch_bounds__(256) void k_attn_scores(
    const float* __restrict__ h, const float* __restrict__ att_w,
    const float* __restrict__ att_b, const float* __restrict__ fc_w,
    float* __restrict__ score, float* __restrict__ dval) {
  const int row = blockIdx.x;
  const int t = threadIdx.x;
  __shared__ float r1[256], r2[256];
  float s1 = 0.f, s2 = 0.f;
#pragma unroll
  for (int j = 0; j < 4; ++j) {
    const int c = t + j * 256;
    const float x = h[(long)row * Hdim + c];
    s1 += x * att_w[c];
    s2 += x * fc_w[c];
  }
  r1[t] = s1; r2[t] = s2;
  __syncthreads();
  for (int st = 128; st > 0; st >>= 1) {
    if (t < st) { r1[t] += r1[t + st]; r2[t] += r2[t + st]; }
    __syncthreads();
  }
  if (t == 0) {
    score[row] = r1[0] + att_b[0];
    dval[row] = r2[0];
  }
}

// ---------- softmax over L + weighted sum + fc bias ----------
__global__ __launch_bounds__(256) void k_attn_final(
    const float* __restrict__ score, const float* __restrict__ dval,
    const float* __restrict__ fc_b, float* __restrict__ out) {
  const int b = blockIdx.x;
  const int t = threadIdx.x;
  __shared__ float red[256];
  float m = -1e30f;
  for (int l = t; l < Ldim; l += 256) m = fmaxf(m, score[b * Ldim + l]);
  red[t] = m;
  __syncthreads();
  for (int st = 128; st > 0; st >>= 1) {
    if (t < st) red[t] = fmaxf(red[t], red[t + st]);
    __syncthreads();
  }
  m = red[0];
  __syncthreads();
  float se = 0.f, sd = 0.f;
  for (int l = t; l < Ldim; l += 256) {
    const float e = expf(score[b * Ldim + l] - m);
    se += e;
    sd += e * dval[b * Ldim + l];
  }
  red[t] = se;
  __syncthreads();
  for (int st = 128; st > 0; st >>= 1) {
    if (t < st) red[t] += red[t + st];
    __syncthreads();
  }
  se = red[0];
  __syncthreads();
  red[t] = sd;
  __syncthreads();
  for (int st = 128; st > 0; st >>= 1) {
    if (t < st) red[t] += red[t + st];
    __syncthreads();
  }
  if (t == 0) out[b] = red[0] / se + fc_b[0];
}

// =======================================================================
extern "C" void kernel_launch(void* const* d_in, const int* in_sizes, int n_in,
                              void* d_out, int out_size, void* d_ws, size_t ws_size,
                              hipStream_t stream) {
  (void)in_sizes; (void)n_in; (void)out_size; (void)ws_size;

  const float* x = (const float*)d_in[0];
  const float* conv_w[3]  = {(const float*)d_in[3], (const float*)d_in[7],  (const float*)d_in[11]};
  const float* conv_b[3]  = {(const float*)d_in[4], (const float*)d_in[8],  (const float*)d_in[12]};
  const float* conv_g[3]  = {(const float*)d_in[5], (const float*)d_in[9],  (const float*)d_in[13]};
  const float* conv_be[3] = {(const float*)d_in[6], (const float*)d_in[10], (const float*)d_in[14]};
  const float* s4_log_dt = (const float*)d_in[15];
  const float* s4_logA   = (const float*)d_in[16];
  const float* s4_Aim    = (const float*)d_in[17];
  const float* s4_Cre    = (const float*)d_in[18];
  const float* s4_Cim    = (const float*)d_in[19];
  const float* s4_D      = (const float*)d_in[20];
  const float* s4_ow     = (const float*)d_in[21];
  const float* s4_ob     = (const float*)d_in[22];
  const float* s4_g      = (const float*)d_in[23];
  const float* s4_be     = (const float*)d_in[24];
  const float* cbam_w1   = (const float*)d_in[25];
  const float* cbam_b1   = (const float*)d_in[26];
  const float* cbam_w2   = (const float*)d_in[27];
  const float* cbam_b2   = (const float*)d_in[28];
  const float* cbam_spw  = (const float*)d_in[29];
  const float* cbam_spb  = (const float*)d_in[30];
  const float* att_w     = (const float*)d_in[31];
  const float* att_b     = (const float*)d_in[32];
  const float* fc_w      = (const float*)d_in[33];
  const float* fc_b      = (const float*)d_in[34];
  float* out = (float*)d_out;

  // ---- workspace carve (bump allocator, 256B aligned) ----
  char* ws = (char*)d_ws;
  size_t off = 0;
  auto carve = [&](size_t bytes) -> void* {
    void* p = ws + off;
    off = (off + bytes + 255) & ~(size_t)255;
    return p;
  };
  const size_t nBLH = (size_t)Mrows * Hdim;              // 8M elements
  float*          hbuf   = (float*)carve(nBLH * 4);
  float*          ybuf   = (float*)carve(nBLH * 4);
  float*          y2buf  = (float*)carve(nBLH * 2 * 4);
  unsigned short* abuf   = (unsigned short*)carve(nBLH * 2);
  unsigned short* wbuf   = (unsigned short*)carve((size_t)11 * Hdim * Hdim * 2);
  float* cWr   = (float*)carve(Hdim * N2s * 4);
  float* cWi   = (float*)carve(Hdim * N2s * 4);
  float* cCr   = (float*)carve(Hdim * N2s * 4);
  float* cCi   = (float*)carve(Hdim * N2s * 4);
  float* avg   = (float*)carve(Bdim * Hdim * 4);
  float* mxp   = (float*)carve(Bdim * Hdim * 4);
  float* gate  = (float*)carve(Bdim * Hdim * 4);
  float* spmean= (float*)carve(Mrows * 4);
  float* spmax = (float*)carve(Mrows * 4);
  float* score = (float*)carve(Mrows * 4);
  float* dval  = (float*)carve(Mrows * 4);

  const int nElem = (int)nBLH;                  // 8388608
  const dim3 blk(256);
  const int ks[3] = {3, 7, 11};

  // ---- conv stack ----
  k_cast_bf16<<<(nElem + 255) / 256, blk, 0, stream>>>(x, abuf, nElem);
  for (int ci = 0; ci < 3; ++ci) {
    if (ci > 0)
      k_cast_bf16<<<(nElem + 255) / 256, blk, 0, stream>>>(hbuf, abuf, nElem);
    const int ksz = ks[ci];
    const int nw = ksz * Hdim * Hdim;
    k_prep_convw<<<(nw + 255) / 256, blk, 0, stream>>>(conv_w[ci], wbuf, ksz, nw);
    dim3 grid(Hdim / 256, Mrows / 32);
    k_gemm_bf16<<<grid, blk, 0, stream>>>(abuf, wbuf, conv_b[ci], ybuf, Hdim, ksz);
    k_cln_relu<<<Mrows, blk, 0, stream>>>(ybuf, conv_g[ci], conv_be[ci], hbuf);
  }

  // ---- CBAM ----
  k_cbam_pool<<<(Bdim * Hdim) / 256, blk, 0, stream>>>(hbuf, avg, mxp);
  k_cbam_mlp<<<Bdim, 64, 0, stream>>>(avg, mxp, cbam_w1, cbam_b1, cbam_w2, cbam_b2, gate);
  k_cbam_scale_sp<<<Mrows, blk, 0, stream>>>(hbuf, gate, spmean, spmax);
  k_cbam_spatial<<<Mrows, blk, 0, stream>>>(hbuf, spmean, spmax, cbam_spw, cbam_spb);

  // ---- S4 blocks ----
  for (int i = 0; i < NSSM; ++i) {
    k_s4_coef<<<(Hdim * N2s) / 256, blk, 0, stream>>>(
        s4_log_dt + i * Hdim, s4_logA + (size_t)i * Hdim * N2s,
        s4_Aim + (size_t)i * Hdim * N2s, s4_Cre + (size_t)i * Hdim * N2s,
        s4_Cim + (size_t)i * Hdim * N2s, cWr, cWi, cCr, cCi);
    k_s4_scan<<<(Bdim * Hdim) / 256, blk, 0, stream>>>(
        hbuf, cWr, cWi, cCr, cCi, s4_D + i * Hdim, abuf);
    const int nw = Hdim * 2 * Hdim;
    k_prep_ow<<<(nw + 255) / 256, blk, 0, stream>>>(
        s4_ow + (size_t)i * 2 * Hdim * Hdim, wbuf, nw);
    dim3 grid(2 * Hdim / 256, Mrows / 32);
    k_gemm_bf16<<<grid, blk, 0, stream>>>(abuf, wbuf, s4_ob + i * 2 * Hdim,
                                          y2buf, 2 * Hdim, 1);
    k_glu<<<(nElem + 255) / 256, blk, 0, stream>>>(y2buf, ybuf, nElem);
    k_cln_relu<<<Mrows, blk, 0, stream>>>(ybuf, s4_g + i * Hdim, s4_be + i * Hdim, hbuf);
  }

  // ---- attention pooling + fc ----
  k_attn_scores<<<Mrows, blk, 0, stream>>>(hbuf, att_w, att_b, fc_w, score, dval);
  k_attn_final<<<Bdim, blk, 0, stream>>>(score, dval, fc_b, out);
}
